// DeepGCN_66520453480972
// MI455X (gfx1250) — matrix operand order repaired and checked by reading, not compiled, not run on hardware
//
#include <hip/hip_runtime.h>

typedef __attribute__((ext_vector_type(16))) _Float16 v16h;
typedef __attribute__((ext_vector_type(8)))  _Float16 v8h;
typedef __attribute__((ext_vector_type(8)))  float    v8f;

#define B_   8
#define N_   4096
#define C_   64
#define BN_  (B_*N_)
#define KNB  16
#define EPSBN 1e-5f
#define FINF  3.0e38f

__device__ __forceinline__ v16h pack16(v8h lo, v8h hi){
  v16h r;
#pragma unroll
  for(int e=0;e<8;++e){ r[e]=lo[e]; r[e+8]=hi[e]; }
  return r;
}
__device__ __forceinline__ v8f wmma_f16(v16h a, v16h b, v8f c){
  return __builtin_amdgcn_wmma_f32_16x16x32_f16(false,a,false,b,(short)0,c,false,false);
}
// wave-internal LDS producer->consumer ordering: DS ops of a wave are in-order,
// so an explicit split-counter wait + compiler memory barrier is sufficient
// (no block-wide s_barrier needed for wave-private staging buffers).
__device__ __forceinline__ void lds_wave_fence(){
  asm volatile("s_wait_dscnt 0" ::: "memory");
}

// ---------------- weight conversion f32 -> f16 ----------------
__global__ void k_convert_w(const float* wb, const float* wf, _Float16* wb16, _Float16* wf16){
  int i = blockIdx.x*blockDim.x + threadIdx.x;
  const int nb = 6*C_*2*C_;   // 49152
  const int nf = 256*7*C_;    // 114688
  if(i < nb)           wb16[i]      = (_Float16)wb[i];
  else if(i < nb+nf)   wf16[i-nb]   = (_Float16)wf[i-nb];
}

// ---------------- pad raw coords to K=32 f16 + sq norms ----------------
__global__ void k_prep_x(const float* x, _Float16* xpad, float* sq){
  int n = blockIdx.x*blockDim.x + threadIdx.x;
  if(n>=BN_) return;
  float a=x[n*3],b=x[n*3+1],c=x[n*3+2];
  sq[n]=a*a+b*b+c*c;
  _Float16* o = xpad + (size_t)n*32;
#pragma unroll
  for(int j=0;j<32;++j) o[j]=(_Float16)0.f;
  o[0]=(_Float16)a; o[1]=(_Float16)b; o[2]=(_Float16)c;
}

// ---------------- per-point squared norm over C_ channels ----------------
__global__ void k_sq(const float* f, float* sq){
  int n = blockIdx.x*blockDim.x + threadIdx.x;
  if(n>=BN_) return;
  const float* r=f+(size_t)n*C_;
  float s=0.f;
#pragma unroll
  for(int c=0;c<C_;++c){ float v=r[c]; s+=v*v; }
  sq[n]=s;
}

// ---------------- fused WMMA KNN + top-(k*dil) selection ----------------
// one workgroup = 16 query rows of one batch; 4 independent waves, each sweeps
// 32 iterations x (2 column tiles) with no inter-wave sync until the merge.
template<int CD>
__global__ void __launch_bounds__(128) k_knn(const _Float16* __restrict__ feat,
                                             const float* __restrict__ sq,
                                             int* __restrict__ idxout, int dil){
  __shared__ float listD[4*16*96];
  __shared__ int   listI[4*16*96];
  __shared__ float stage[4][16][32];
  __shared__ float sqA[16];
  const int keep = KNB*dil;               // <= 96
  const int b    = blockIdx.x / (N_/16);
  const int row0 = (blockIdx.x % (N_/16))*16;
  const int tx   = threadIdx.x;
  const int wave = tx>>5, lane = tx&31;
  const int m    = lane&15, hi = lane>>4;

  for(int i=tx;i<4*16*96;i+=128){ listD[i]=FINF; listI[i]=0; }
  if(tx<16) sqA[tx]=sq[b*N_+row0+tx];
  __syncthreads();

  // A fragments (rows of the query tile), CD/32 k-steps, kept in VGPRs
  v16h a[CD/32];
  {
    const _Float16* arow = feat + ((size_t)(b*N_+row0+m))*CD;
#pragma unroll
    for(int s=0;s<CD/32;++s){
      v8h lo = *(const v8h*)(arow + 32*s + (hi?16:0));
      v8h h8 = *(const v8h*)(arow + 32*s + (hi?24:8));
      a[s]=pack16(lo,h8);
    }
  }

  float thr = FINF;                       // per-row threshold (lanes 0..15)
  const _Float16* featB0 = feat + ((size_t)b*N_)*CD;

  for(int t2=wave; t2<N_/32; t2+=4){
    const int col0 = t2*32;
    // prefetch next iteration's B rows into cache (global_prefetch_b8)
    {
      int nt = t2+4; if(nt>=N_/32) nt=t2;
      __builtin_prefetch(featB0 + ((size_t)(nt*32+m))*CD, 0, 1);
      __builtin_prefetch(featB0 + ((size_t)(nt*32+16+m))*CD, 0, 1);
    }
    v8f acc0 = {0.f,0.f,0.f,0.f,0.f,0.f,0.f,0.f};
    v8f acc1 = {0.f,0.f,0.f,0.f,0.f,0.f,0.f,0.f};
    const _Float16* brow0 = featB0 + ((size_t)(col0+m))*CD;
    const _Float16* brow1 = featB0 + ((size_t)(col0+16+m))*CD;
#pragma unroll
    for(int s=0;s<CD/32;++s){
      v16h bf0 = *(const v16h*)(brow0 + 32*s + hi*16);
      v16h bf1 = *(const v16h*)(brow1 + 32*s + hi*16);
      acc0 = wmma_f16(a[s], bf0, acc0);
      acc1 = wmma_f16(a[s], bf1, acc1);
    }
    float sb0 = sq[b*N_+col0+m];
    float sb1 = sq[b*N_+col0+16+m];
#pragma unroll
    for(int r=0;r<8;++r){
      stage[wave][r+8*hi][m]    = sqA[r+8*hi] + sb0 - 2.0f*acc0[r];
      stage[wave][r+8*hi][16+m] = sqA[r+8*hi] + sb1 - 2.0f*acc1[r];
    }
    lds_wave_fence();                    // wave-private staging: no s_barrier
    if(lane<16){
      float* LD = listD + (wave*16+lane)*96;
      int*   LI = listI + (wave*16+lane)*96;
      for(int c=0;c<32;++c){
        float d = stage[wave][lane][c];
        if(d < thr){
          int p = keep-1;
          while(p>0 && LD[p-1]>d){ LD[p]=LD[p-1]; LI[p]=LI[p-1]; --p; }
          LD[p]=d; LI[p]=col0+c;
          thr = LD[keep-1];
        }
      }
    }
    asm volatile("" ::: "memory");       // keep next stores after these loads
  }
  __syncthreads();                       // single block barrier before merge

  // merge 4 sorted wave-lists per row, emit every dil-th index
  if(tx<16){
    int pr[4]={0,0,0,0};
    for(int kk=0;kk<keep;++kk){
      float best=FINF; int bw=0;
#pragma unroll
      for(int w=0;w<4;++w){
        float d=listD[(w*16+tx)*96+pr[w]];
        if(d<best){best=d;bw=w;}
      }
      int bi = listI[(bw*16+tx)*96+pr[bw]];
      pr[bw]++;
      if(kk % dil == 0)
        idxout[((size_t)(b*N_+row0+tx))*KNB + kk/dil] = bi;
    }
  }
}

// ---------------- head MRConv (3 -> 64), scalar (K=6 too small for WMMA) ----------------
__global__ void k_head_y(const float* __restrict__ x, const float* __restrict__ wh,
                         const int* __restrict__ idx, float* __restrict__ y){
  int t = blockIdx.x*blockDim.x + threadIdx.x;
  int n = t>>6, o = t&63;
  if(n>=BN_) return;
  int b = n / N_;
  const float* xr = x + (size_t)n*3;
  float x0=xr[0],x1=xr[1],x2=xr[2];
  float m0=-FINF,m1=-FINF,m2=-FINF;
  const int* ir = idx + (size_t)n*KNB;
#pragma unroll
  for(int k=0;k<KNB;++k){
    const float* xn = x + ((size_t)(b*N_+ir[k]))*3;
    m0=fmaxf(m0,xn[0]-x0); m1=fmaxf(m1,xn[1]-x1); m2=fmaxf(m2,xn[2]-x2);
  }
  const float* w = wh + o*6;
  y[(size_t)n*C_+o] = x0*w[0]+x1*w[1]+x2*w[2]+m0*w[3]+m1*w[4]+m2*w[5];
}

// ---------------- BatchNorm: coalesced two-stage reduce (deterministic) ----------------
// stage 1: 128 blocks, each reads 256 fully-contiguous rows; thread address is
// base + tx*4 -> perfectly coalesced. Partials to ws (no atomics).
__global__ void k_bn_reduce(const float* __restrict__ y, int Cch,
                            float* __restrict__ part1, float* __restrict__ part2){
  __shared__ float s1[256], s2[256];
  const int tx  = threadIdx.x;
  const int ch  = tx % Cch;          // Cch in {64,256}, divides 256
  const int rl  = tx / Cch;
  const int rpl = 256 / Cch;
  const size_t row0 = (size_t)blockIdx.x * 256;
  float a=0.f,b2=0.f;
  for(int r=rl; r<256; r+=rpl){
    float v = y[(row0+r)*Cch + ch];
    a+=v; b2+=v*v;
  }
  s1[tx]=a; s2[tx]=b2; __syncthreads();
  if(rl==0){
    for(int g=1;g<rpl;++g){ a+=s1[g*Cch+ch]; b2+=s2[g*Cch+ch]; }
    part1[(size_t)blockIdx.x*Cch+ch]=a;
    part2[(size_t)blockIdx.x*Cch+ch]=b2;
  }
}

// stage 2: fold 128 partials per channel + finalize scale/shift
__global__ void k_bn_fin(const float* __restrict__ part1, const float* __restrict__ part2,
                         const float* __restrict__ g, const float* __restrict__ bet,
                         float* __restrict__ sc, float* __restrict__ sh, int Cch){
  int ch = threadIdx.x;
  if(ch>=Cch) return;
  float s=0.f, ss=0.f;
  for(int i=0;i<128;++i){ s+=part1[(size_t)i*Cch+ch]; ss+=part2[(size_t)i*Cch+ch]; }
  float mu  = s/(float)BN_;
  float var = ss/(float)BN_ - mu*mu;
  float sc0 = g[ch]*rsqrtf(var+EPSBN);
  sc[ch]=sc0; sh[ch]=bet[ch]-mu*sc0;
}

__global__ void k_apply(const float* __restrict__ y, const float* __restrict__ prev,
                        const float* __restrict__ sc, const float* __restrict__ sh,
                        float* __restrict__ outf, _Float16* __restrict__ out16){
  int i = blockIdx.x*blockDim.x + threadIdx.x;
  if(i>=BN_*C_) return;
  int ch = i & (C_-1);
  float v = fmaxf(y[i]*sc[ch]+sh[ch], 0.f);
  if(prev) v += prev[i];
  outf[i]=v; out16[i]=(_Float16)v;
}

// ---------------- MRConv block GEMM (128 -> 64) via WMMA ----------------
__global__ void __launch_bounds__(128) k_mrconv_gemm(const float* __restrict__ f,
                                                     const int* __restrict__ idx,
                                                     const _Float16* __restrict__ W16,
                                                     float* __restrict__ y){
  __shared__ __align__(32) _Float16 h16s[16][128];
  __shared__ __align__(32) _Float16 w16s[64][128];
  const int b    = blockIdx.x/(N_/16);
  const int row0 = (blockIdx.x%(N_/16))*16;
  const int tx   = threadIdx.x;
  for(int i=tx;i<1024;i+=128)
    ((uint4*)&w16s[0][0])[i] = ((const uint4*)W16)[i];
  {
    int r = tx>>3, c0 = (tx&7)*8;
    size_t nrow = (size_t)(b*N_) + row0 + r;
    const float* xr = f + nrow*C_ + c0;
    float xv[8], mv[8];
#pragma unroll
    for(int j=0;j<8;++j){ xv[j]=xr[j]; mv[j]=-FINF; }
    const int* ir = idx + nrow*KNB;
    for(int k=0;k<KNB;++k){
      const float* fn = f + ((size_t)(b*N_)+ir[k])*C_ + c0;
#pragma unroll
      for(int j=0;j<8;++j) mv[j]=fmaxf(mv[j], fn[j]-xv[j]);
    }
#pragma unroll
    for(int j=0;j<8;++j){
      h16s[r][c0+j]    = (_Float16)xv[j];
      h16s[r][64+c0+j] = (_Float16)mv[j];
    }
  }
  __syncthreads();
  const int wave=tx>>5, lane=tx&31, m=lane&15, hi=lane>>4;
  v8f acc = {0.f,0.f,0.f,0.f,0.f,0.f,0.f,0.f};
#pragma unroll
  for(int s=0;s<4;++s){
    v8h lo = *(const v8h*)&h16s[m][32*s+(hi?16:0)];
    v8h h8 = *(const v8h*)&h16s[m][32*s+(hi?24:8)];
    v16h av = pack16(lo,h8);
    v16h bf = *(const v16h*)&w16s[wave*16+m][32*s+(hi?16:0)];
    acc = wmma_f16(av,bf,acc);
  }
#pragma unroll
  for(int r=0;r<8;++r)
    y[((size_t)(b*N_)+row0+r+8*hi)*C_ + wave*16 + m] = acc[r];
}

// ---------------- fusion GEMM (448 -> 256) via WMMA ----------------
__global__ void __launch_bounds__(256) k_fusion_gemm(const _Float16* __restrict__ feat16,
                                                     const _Float16* __restrict__ wf16,
                                                     float* __restrict__ fy){
  __shared__ __align__(32) _Float16 h16s[16][448];
  const int b    = blockIdx.x/(N_/16);
  const int row0 = (blockIdx.x%(N_/16))*16;
  const int tx   = threadIdx.x;
  for(int i=tx;i<896;i+=256){
    int row = i/56;
    int pos = (i%56)*8;
    int lvl = pos>>6;
    int co  = pos&63;
    const _Float16* src = feat16 + (((size_t)lvl*BN_) + b*N_ + row0 + row)*C_ + co;
    *(uint4*)&h16s[row][pos] = *(const uint4*)src;
  }
  __syncthreads();
  const int wave=tx>>5, lane=tx&31, m=lane&15, hi=lane>>4;
  v8f acc0 = {0.f,0.f,0.f,0.f,0.f,0.f,0.f,0.f};
  v8f acc1 = {0.f,0.f,0.f,0.f,0.f,0.f,0.f,0.f};
#pragma unroll
  for(int s=0;s<14;++s){
    v8h lo=*(const v8h*)&h16s[m][32*s+(hi?16:0)];
    v8h h8=*(const v8h*)&h16s[m][32*s+(hi?24:8)];
    v16h av=pack16(lo,h8);
    const _Float16* b0=wf16 + ((size_t)(wave*16+m))*448     + 32*s + hi*16;
    const _Float16* b1=wf16 + ((size_t)((wave+8)*16+m))*448 + 32*s + hi*16;
    acc0 = wmma_f16(av, *(const v16h*)b0, acc0);
    acc1 = wmma_f16(av, *(const v16h*)b1, acc1);
  }
#pragma unroll
  for(int r=0;r<8;++r){
    size_t base = ((size_t)(b*N_)+row0+r+8*hi)*256;
    fy[base + wave*16+m]      = acc0[r];
    fy[base + (wave+8)*16+m]  = acc1[r];
  }
}

// ---------------- coalesced two-stage BN+leaky + max/mean pool ----------------
__global__ void k_pool1(const float* __restrict__ fy, const float* __restrict__ sc,
                        const float* __restrict__ sh,
                        float* __restrict__ pmax, float* __restrict__ psum){
  const int b = blockIdx.x>>4, chunk = blockIdx.x&15;
  const int ch = threadIdx.x;
  const size_t row0 = (size_t)b*N_ + chunk*256;
  const float s=sc[ch], h=sh[ch];
  float mx=-FINF, sm=0.f;
  for(int r=0;r<256;++r){
    float v = fy[(row0+r)*256+ch]*s+h;
    v = v>0.f? v : 0.2f*v;
    mx = fmaxf(mx,v); sm += v;
  }
  pmax[(size_t)blockIdx.x*256+ch]=mx;
  psum[(size_t)blockIdx.x*256+ch]=sm;
}

__global__ void k_pool2(const float* __restrict__ pmax, const float* __restrict__ psum,
                        float* __restrict__ p){
  const int b = blockIdx.x, ch = threadIdx.x;
  float mx=-FINF, sm=0.f;
  for(int c2=0;c2<16;++c2){
    mx = fmaxf(mx, pmax[((size_t)b*16+c2)*256+ch]);
    sm += psum[((size_t)b*16+c2)*256+ch];
  }
  p[(size_t)b*512+ch]     = mx;
  p[(size_t)b*512+256+ch] = sm/(float)N_;
}

// ---------------- final linear + BN over batch + leaky ----------------
__global__ void k_merge(const float* __restrict__ p, const float* __restrict__ wm,
                        const float* __restrict__ bias, const float* __restrict__ g,
                        const float* __restrict__ bet, float* __restrict__ out){
  int o = threadIdx.x;
  float yv[8];
  const float* w = wm + (size_t)o*512;
  float s=0.f;
  for(int b=0;b<8;++b){
    float acc = bias[o];
    const float* pb = p + (size_t)b*512;
    for(int j=0;j<512;++j) acc += pb[j]*w[j];
    yv[b]=acc; s+=acc;
  }
  float mu = s*0.125f;
  float var=0.f;
#pragma unroll
  for(int b=0;b<8;++b){ float d=yv[b]-mu; var+=d*d; }
  var*=0.125f;
  float sc = g[o]*rsqrtf(var+EPSBN);
#pragma unroll
  for(int b=0;b<8;++b){
    float v=(yv[b]-mu)*sc + bet[o];
    out[b*256+o] = v>0.f? v : 0.2f*v;
  }
}

extern "C" void kernel_launch(void* const* d_in, const int* in_sizes, int n_in,
                              void* d_out, int out_size, void* d_ws, size_t ws_size,
                              hipStream_t stream){
  const float* x        = (const float*)d_in[0];
  const float* w_head   = (const float*)d_in[1];
  const float* g_head   = (const float*)d_in[2];
  const float* b_head   = (const float*)d_in[3];
  const float* w_blocks = (const float*)d_in[4];
  const float* g_blocks = (const float*)d_in[5];
  const float* b_blocks = (const float*)d_in[6];
  const float* w_fusion = (const float*)d_in[7];
  const float* g_fusion = (const float*)d_in[8];
  const float* b_fusion = (const float*)d_in[9];
  const float* w_merge  = (const float*)d_in[10];
  const float* bias_m   = (const float*)d_in[11];
  const float* g_merge  = (const float*)d_in[12];
  const float* b_merge  = (const float*)d_in[13];

  char* w = (char*)d_ws;
  size_t off=0;
  auto alloc=[&](size_t bytes)->char*{
    char* r=w+off; off=(off+bytes+255)&~(size_t)255; return r;
  };
  _Float16* feat16 = (_Float16*)alloc((size_t)7*BN_*C_*2);
  _Float16* xpad16 = (_Float16*)alloc((size_t)BN_*32*2);
  float*    featA  = (float*)alloc((size_t)BN_*C_*4);
  float*    featB  = (float*)alloc((size_t)BN_*C_*4);
  float*    ybuf   = (float*)alloc((size_t)BN_*C_*4);
  int*      idxb   = (int*)alloc((size_t)BN_*KNB*4);
  float*    sqb    = (float*)alloc((size_t)BN_*4);
  float*    part1  = (float*)alloc((size_t)128*256*4);
  float*    part2  = (float*)alloc((size_t)128*256*4);
  float*    bsc    = (float*)alloc(256*4);
  float*    bsh    = (float*)alloc(256*4);
  _Float16* wb16   = (_Float16*)alloc((size_t)6*C_*2*C_*2);
  _Float16* wf16   = (_Float16*)alloc((size_t)256*7*C_*2);
  float*    fy     = (float*)alloc((size_t)BN_*256*4);
  float*    pmax   = (float*)alloc((size_t)B_*16*256*4);
  float*    psum   = (float*)alloc((size_t)B_*16*256*4);
  float*    pbuf   = (float*)alloc((size_t)8*512*4);
  float*    outv   = (float*)d_out;

  k_convert_w<<<(49152+114688+255)/256,256,0,stream>>>(w_blocks,w_fusion,wb16,wf16);
  k_prep_x<<<BN_/256,256,0,stream>>>(x,xpad16,sqb);

  // head: KNN on raw coords (dil=1), MRConv 3->64, BN+ReLU
  k_knn<32><<<B_*(N_/16),128,0,stream>>>(xpad16,sqb,idxb,1);
  k_head_y<<<(BN_*64)/256,256,0,stream>>>(x,w_head,idxb,ybuf);
  k_bn_reduce<<<128,256,0,stream>>>(ybuf,C_,part1,part2);
  k_bn_fin<<<1,C_,0,stream>>>(part1,part2,g_head,b_head,bsc,bsh,C_);
  k_apply<<<(BN_*C_)/256,256,0,stream>>>(ybuf,nullptr,bsc,bsh,featA,feat16);

  float* cur=featA; float* nxt=featB;
  for(int i=0;i<6;++i){
    k_sq<<<BN_/256,256,0,stream>>>(cur,sqb);
    k_knn<64><<<B_*(N_/16),128,0,stream>>>(feat16 + (size_t)i*BN_*C_, sqb, idxb, i+1);
    k_mrconv_gemm<<<B_*(N_/16),128,0,stream>>>(cur, idxb, wb16 + (size_t)i*C_*2*C_, ybuf);
    k_bn_reduce<<<128,256,0,stream>>>(ybuf,C_,part1,part2);
    k_bn_fin<<<1,C_,0,stream>>>(part1,part2,g_blocks+(size_t)i*C_, b_blocks+(size_t)i*C_, bsc,bsh,C_);
    k_apply<<<(BN_*C_)/256,256,0,stream>>>(ybuf,cur,bsc,bsh,nxt,feat16+(size_t)(i+1)*BN_*C_);
    float* tmp=cur; cur=nxt; nxt=tmp;
  }

  // fusion 448->256, BN+leaky folded into pooling
  k_fusion_gemm<<<B_*(N_/16),256,0,stream>>>(feat16, wf16, fy);
  k_bn_reduce<<<128,256,0,stream>>>(fy,256,part1,part2);
  k_bn_fin<<<1,256,0,stream>>>(part1,part2,g_fusion,b_fusion,bsc,bsh,256);
  k_pool1<<<B_*16,256,0,stream>>>(fy,bsc,bsh,pmax,psum);
  k_pool2<<<B_,256,0,stream>>>(pmax,psum,pbuf);
  k_merge<<<1,256,0,stream>>>(pbuf,w_merge,bias_m,g_merge,b_merge,outv);

  (void)in_sizes;(void)n_in;(void)out_size;(void)ws_size;
}
